// GPT2AttentionSmoothQuant_31413390802944
// MI455X (gfx1250) — 16-bit and 8-bit matrix operand order repaired and checked by reading, compiled, not run on hardware
//
#include <hip/hip_runtime.h>
#include <stdint.h>

#define BB  2
#define SS  2048
#define EE  2048
#define HH  16
#define HD  128
#define O3  (3 * EE)
#define BS  (BB * SS)

typedef __attribute__((ext_vector_type(16))) _Float16 v16h;
typedef __attribute__((ext_vector_type(8)))  float    v8f;
typedef __attribute__((ext_vector_type(8)))  int      v8i;
typedef __attribute__((ext_vector_type(4)))  int      v4i;
typedef __attribute__((ext_vector_type(4)))  unsigned v4u;

union I8Frag  { v8i v; int i[8]; int2 i2[4]; int4 i4[2]; };
union H16Frag { v16h v; _Float16 h[16]; unsigned u[8]; };

__device__ __forceinline__ void tdm_copy_to_lds(const void* gsrc, unsigned lds_off,
                                                unsigned bytes)
{
    const unsigned units = bytes >> 3;
    const unsigned long long ga = (unsigned long long)(uintptr_t)gsrc;

    v4u g0;
    g0[0] = 1u;
    g0[1] = lds_off;
    g0[2] = (unsigned)(ga & 0xFFFFFFFFu);
    g0[3] = (unsigned)((ga >> 32) & 0x1FFFFFFu) | (2u << 30);

    v8i g1;
    g1[0] = (int)(3u << 16);
    g1[1] = (int)(units << 16);
    g1[2] = (int)(((units >> 16) & 0xFFFFu) | (1u << 16));
    g1[3] = (int)(units << 16);
    g1[4] = 1;
    g1[5] = (int)units;
    g1[6] = 0;
    g1[7] = 0;

    const v4i z4 = {};
#if __clang_major__ >= 23
    const v8i z8 = {};
    __builtin_amdgcn_tensor_load_to_lds(g0, g1, z4, z4, z8, 0);
#else
    __builtin_amdgcn_tensor_load_to_lds(g0, g1, z4, z4, 0);
#endif
}

__global__ __launch_bounds__(256)
void qkv_gemm_kernel(const int8_t* __restrict__ hs, const int8_t* __restrict__ W,
                     const float* __restrict__ bias, const float* __restrict__ wscale,
                     const float* __restrict__ sact,
                     _Float16* __restrict__ Qf, _Float16* __restrict__ Kf,
                     _Float16* __restrict__ Vf)
{
    __shared__ __align__(16) int8_t lds_a[16 * EE];

    const int lane = threadIdx.x & 31;
    const int w    = threadIdx.x >> 5;
    const int hi   = lane >> 4;
    const int lc   = lane & 15;
    const int m0   = blockIdx.x * 16;
    const int n0   = (blockIdx.y * 8 + w) * 16;

    if (w == 0) {
        tdm_copy_to_lds(hs + (size_t)m0 * EE, (unsigned)(uintptr_t)lds_a, 16 * EE);
        __builtin_amdgcn_s_wait_tensorcnt(0);
    }
    __syncthreads();

    const int8_t* arow = lds_a + lc * EE;
    const int8_t* brow = W + (size_t)(n0 + lc) * EE;

    I8Frag acc; acc.v = (v8i){};

    for (int k0 = 0; k0 < EE; k0 += 64) {
        I8Frag a, b;
        #pragma unroll
        for (int p = 0; p < 4; ++p)
            a.i2[p] = *(const int2*)(arow + k0 + (p >> 1) * 32 + hi * 16 + (p & 1) * 8);
        b.i4[0] = *(const int4*)(brow + k0 + hi * 16);
        b.i4[1] = *(const int4*)(brow + k0 + 32 + hi * 16);

        __builtin_prefetch(brow + k0 + 256, 0, 3);

        acc.v = __builtin_amdgcn_wmma_i32_16x16x64_iu8(
            true, a.v, true, b.v, acc.v, false, false);
    }

    const float sa = sact[0];
    const int   o  = n0 + lc;
    const float ws = sa * wscale[o];
    const float bo = bias[o];
    const int sect = o >> 11;
    const int oo   = o & (EE - 1);
    const int hh   = oo >> 7;
    const int d    = oo & (HD - 1);
    _Float16* dst  = (sect == 0) ? Qf : (sect == 1) ? Kf : Vf;

    #pragma unroll
    for (int r = 0; r < 8; ++r) {
        const int tok = m0 + hi * 8 + r;
        const int bb  = tok >> 11;
        const int s   = tok & (SS - 1);
        const float val = (float)acc.i[r] * ws + bo;
        dst[(((size_t)bb * HH + hh) * SS + s) * HD + d] = (_Float16)val;
    }
}

__global__ __launch_bounds__(256)
void attn_kernel(const _Float16* __restrict__ Qf, const _Float16* __restrict__ Kf,
                 const _Float16* __restrict__ Vf, float* __restrict__ ctx)
{
    __shared__ __align__(16) _Float16 pbuf[8][16][32];

    const int lane = threadIdx.x & 31;
    const int w    = threadIdx.x >> 5;
    const int hi   = lane >> 4;
    const int lc   = lane & 15;

    const int g  = blockIdx.x * 8 + w;
    const int bh = g >> 7;
    const int m0 = (g & 127) * 16;
    const int b  = bh >> 4;
    const int h  = bh & 15;

    const _Float16* Qb = Qf + (size_t)bh * SS * HD;
    const _Float16* Kb = Kf + (size_t)bh * SS * HD;
    const _Float16* Vb = Vf + (size_t)bh * SS * HD;

    H16Frag qf[4];
    {
        const _Float16* qrow = Qb + (size_t)(m0 + lc) * HD;
        #pragma unroll
        for (int c = 0; c < 4; ++c)
            #pragma unroll
            for (int v = 0; v < 8; ++v) {
                const int kb = hi * 16 + v * 2;
                qf[c].u[v] = *(const unsigned*)(qrow + c * 32 + kb);
            }
    }

    const v8f zf = {};
    v8f ctxa[8];
    #pragma unroll
    for (int t = 0; t < 8; ++t) ctxa[t] = zf;
    float rowmax[8], lsum[8];
    #pragma unroll
    for (int r = 0; r < 8; ++r) { rowmax[r] = -1e30f; lsum[r] = 0.0f; }

    const float sc = 0.08838834764831845f;

    for (int kk = 0; kk < m0 + 16; kk += 32) {
        v8f s0 = zf, s1 = zf;
        #pragma unroll
        for (int c = 0; c < 4; ++c) {
            H16Frag kf0, kf1;
            const unsigned* kp0 =
                (const unsigned*)(Kb + (size_t)(kk + lc) * HD + c * 32 + hi * 16);
            const unsigned* kp1 =
                (const unsigned*)(Kb + (size_t)(kk + 16 + lc) * HD + c * 32 + hi * 16);
            #pragma unroll
            for (int v = 0; v < 8; ++v) { kf0.u[v] = kp0[v]; kf1.u[v] = kp1[v]; }
            s0 = __builtin_amdgcn_wmma_f32_16x16x32_f16(
                false, qf[c].v, false, kf0.v, (short)0, s0, false, false);
            s1 = __builtin_amdgcn_wmma_f32_16x16x32_f16(
                false, qf[c].v, false, kf1.v, (short)0, s1, false, false);
        }

        const bool need_mask = (kk + 31 > m0);
        float p0[8], p1[8], corr[8];
        #pragma unroll
        for (int r = 0; r < 8; ++r) {
            float a0 = s0[r] * sc, a1 = s1[r] * sc;
            if (need_mask) {
                const int qrow = m0 + hi * 8 + r;
                if (kk + lc > qrow)      a0 = -1e30f;
                if (kk + 16 + lc > qrow) a1 = -1e30f;
            }
            float mloc = fmaxf(a0, a1);
            #pragma unroll
            for (int m = 1; m < 16; m <<= 1)
                mloc = fmaxf(mloc, __shfl_xor(mloc, m, 16));
            const float mnew = fmaxf(rowmax[r], mloc);
            corr[r]   = __expf(rowmax[r] - mnew);
            rowmax[r] = mnew;
            p0[r] = __expf(a0 - mnew);
            p1[r] = __expf(a1 - mnew);
            float rs = p0[r] + p1[r];
            #pragma unroll
            for (int m = 1; m < 16; m <<= 1)
                rs += __shfl_xor(rs, m, 16);
            lsum[r] = lsum[r] * corr[r] + rs;
        }
        #pragma unroll
        for (int t = 0; t < 8; ++t)
            #pragma unroll
            for (int r = 0; r < 8; ++r)
                ctxa[t][r] *= corr[r];

        #pragma unroll
        for (int r = 0; r < 8; ++r) {
            const int row = hi * 8 + r;
            pbuf[w][row][lc]      = (_Float16)p0[r];
            pbuf[w][row][lc + 16] = (_Float16)p1[r];
        }
        H16Frag pf;
        #pragma unroll
        for (int v = 0; v < 8; ++v) {
            const int kb = hi * 16 + v * 2;
            pf.u[v] = *(const unsigned*)&pbuf[w][lc][kb];
        }

        #pragma unroll
        for (int t = 0; t < 8; ++t) {
            H16Frag vf;
            const _Float16* vp =
                Vb + (size_t)(kk + hi * 16) * HD + t * 16 + lc;
            #pragma unroll
            for (int v = 0; v < 8; ++v) {
                vf.h[2 * v]     = vp[(size_t)(2 * v) * HD];
                vf.h[2 * v + 1] = vp[(size_t)(2 * v + 1) * HD];
            }
            ctxa[t] = __builtin_amdgcn_wmma_f32_16x16x32_f16(
                false, pf.v, false, vf.v, (short)0, ctxa[t], false, false);
        }
    }

    #pragma unroll
    for (int r = 0; r < 8; ++r) {
        const float inv = 1.0f / lsum[r];
        const int s = m0 + hi * 8 + r;
        float* orow = ctx + ((size_t)b * SS + s) * EE + h * HD;
        #pragma unroll
        for (int t = 0; t < 8; ++t)
            orow[t * 16 + lc] = ctxa[t][r] * inv;
    }
}

__global__ __launch_bounds__(256)
void quant_kernel(const float* __restrict__ ctx, int8_t* __restrict__ qctx,
                  float* __restrict__ tscale)
{
    __shared__ float red[256];
    __shared__ float ssc;
    const int tok = blockIdx.x;
    const int tid = threadIdx.x;
    const float* row = ctx + (size_t)tok * EE;

    const float4 v0 = *(const float4*)(row + tid * 8);
    const float4 v1 = *(const float4*)(row + tid * 8 + 4);
    float am = fmaxf(fmaxf(fmaxf(fabsf(v0.x), fabsf(v0.y)),
                           fmaxf(fabsf(v0.z), fabsf(v0.w))),
                     fmaxf(fmaxf(fabsf(v1.x), fabsf(v1.y)),
                           fmaxf(fabsf(v1.z), fabsf(v1.w))));
    red[tid] = am;
    __syncthreads();
    for (int s = 128; s > 0; s >>= 1) {
        if (tid < s) red[tid] = fmaxf(red[tid], red[tid + s]);
        __syncthreads();
    }
    if (tid == 0) {
        float scv = red[0] / 127.0f;
        if (!(scv > 1e-30f)) scv = 1e-30f;
        ssc = scv;
        tscale[tok] = scv;
    }
    __syncthreads();
    const float inv = 1.0f / ssc;

    union { int8_t c[8]; int2 p; } q;
    const float vv[8] = { v0.x, v0.y, v0.z, v0.w, v1.x, v1.y, v1.z, v1.w };
    #pragma unroll
    for (int j = 0; j < 8; ++j) {
        float t = rintf(vv[j] * inv);
        t = fminf(fmaxf(t, -128.0f), 127.0f);
        q.c[j] = (int8_t)(int)t;
    }
    *(int2*)(qctx + (size_t)tok * EE + tid * 8) = q.p;
}

__global__ __launch_bounds__(256)
void proj_gemm_kernel(const int8_t* __restrict__ qctx, const int8_t* __restrict__ W,
                      const float* __restrict__ bias, const float* __restrict__ wscale,
                      const float* __restrict__ tscale, float* __restrict__ out)
{
    __shared__ __align__(16) int8_t lds_a[16 * EE];

    const int lane = threadIdx.x & 31;
    const int w    = threadIdx.x >> 5;
    const int hi   = lane >> 4;
    const int lc   = lane & 15;
    const int m0   = blockIdx.x * 16;
    const int n0   = (blockIdx.y * 8 + w) * 16;

    if (w == 0) {
        tdm_copy_to_lds(qctx + (size_t)m0 * EE, (unsigned)(uintptr_t)lds_a, 16 * EE);
        __builtin_amdgcn_s_wait_tensorcnt(0);
    }
    __syncthreads();

    const int8_t* arow = lds_a + lc * EE;
    const int8_t* brow = W + (size_t)(n0 + lc) * EE;

    I8Frag acc; acc.v = (v8i){};

    for (int k0 = 0; k0 < EE; k0 += 64) {
        I8Frag a, b;
        #pragma unroll
        for (int p = 0; p < 4; ++p)
            a.i2[p] = *(const int2*)(arow + k0 + (p >> 1) * 32 + hi * 16 + (p & 1) * 8);
        b.i4[0] = *(const int4*)(brow + k0 + hi * 16);
        b.i4[1] = *(const int4*)(brow + k0 + 32 + hi * 16);

        __builtin_prefetch(brow + k0 + 256, 0, 3);

        acc.v = __builtin_amdgcn_wmma_i32_16x16x64_iu8(
            true, a.v, true, b.v, acc.v, false, false);
    }

    const int   o  = n0 + lc;
    const float ws = wscale[o];
    const float bo = bias[o];
    #pragma unroll
    for (int r = 0; r < 8; ++r) {
        const int tok = m0 + hi * 8 + r;
        const float val = (float)acc.i[r] * (tscale[tok] * ws) + bo;
        out[(size_t)tok * EE + o] = val;
    }
}

extern "C" void kernel_launch(void* const* d_in, const int* in_sizes, int n_in,
                              void* d_out, int out_size, void* d_ws, size_t ws_size,
                              hipStream_t stream)
{
    const int8_t* hs        = (const int8_t*)d_in[0];
    const int8_t* c_attn_w  = (const int8_t*)d_in[1];
    const float*  c_attn_b  = (const float*) d_in[2];
    const int8_t* c_proj_w  = (const int8_t*)d_in[3];
    const float*  c_proj_b  = (const float*) d_in[4];
    const float*  s_attn_o  = (const float*) d_in[5];
    const float*  s_attn_w  = (const float*) d_in[6];
    const float*  s_proj_w  = (const float*) d_in[7];
    float* out = (float*)d_out;

    char* ws = (char*)d_ws;
    _Float16* Qf   = (_Float16*)(ws);
    _Float16* Kf   = (_Float16*)(ws + (size_t)16 * 1048576);
    _Float16* Vf   = (_Float16*)(ws + (size_t)32 * 1048576);
    float*    ctx  = (float*)   (ws + (size_t)48 * 1048576);
    int8_t*   qctx = (int8_t*)  (ws + (size_t)80 * 1048576);
    float*    tsc  = (float*)   (ws + (size_t)88 * 1048576);

    const dim3 blk(256);

    qkv_gemm_kernel<<<dim3(BS / 16, O3 / (16 * 8)), blk, 0, stream>>>(
        hs, c_attn_w, c_attn_b, s_attn_w, s_attn_o, Qf, Kf, Vf);

    attn_kernel<<<dim3((BB * HH * (SS / 16)) / 8), blk, 0, stream>>>(
        Qf, Kf, Vf, ctx);

    quant_kernel<<<dim3(BS), blk, 0, stream>>>(ctx, qctx, tsc);

    proj_gemm_kernel<<<dim3(BS / 16, EE / (16 * 8)), blk, 0, stream>>>(
        qctx, c_proj_w, c_proj_b, s_proj_w, tsc, out);
}
